// MultiHeadAttention_40699110097176
// MI455X (gfx1250) — matrix operand order repaired and checked by reading, not compiled, not run on hardware
//
#include <hip/hip_runtime.h>

#define B_DIM 2
#define E_DIM 1024
#define T_DIM 2048
#define H_DIM 16
#define DH_DIM 64

typedef __bf16 bf16;
typedef __attribute__((ext_vector_type(16))) __bf16 v16bf;
typedef __attribute__((ext_vector_type(8)))  __bf16 v8bf;
typedef __attribute__((ext_vector_type(8)))  float  v8f;
typedef __attribute__((ext_vector_type(4)))  float  v4f;

union AFrag { v16bf v; v8bf h[2]; };

__device__ __forceinline__ v8f wmma_bf16(v16bf a, v16bf b, v8f c) {
  // D(f32 16x16) = A(bf16 16x32) * B(bf16 32x16) + C
  return __builtin_amdgcn_wmma_f32_16x16x32_bf16(false, a, false, b, (short)0, c,
                                                 false, false);
}

// ---------------------------------------------------------------------------
// Projection GEMM: out = W @ (X * mask) + bias
//   W  : [1024,1024] f32 row-major
//   X  : [B,1024,2048] f32 (t contiguous)
//   mask: [B,1,T] f32
// MODE 0/1: write bf16 [B,H,T,DH]   (q / k layout for attention fragments)
// MODE 2  : write bf16 [B,E,T]      (v layout)
// MODE 3  : write f32  [B,E,T]      (final output)
// Block: 256 threads (8 waves), tile M=128 x N=128, K-step 32.
// Double-buffered LDS; global loads for step k+1 overlap the WMMAs of step k.
// ---------------------------------------------------------------------------
template <int MODE>
__global__ __launch_bounds__(256) void proj_kernel(
    const float* __restrict__ W, const float* __restrict__ X,
    const float* __restrict__ mask, const float* __restrict__ bias,
    bf16* __restrict__ outB, float* __restrict__ outF) {
  __shared__ bf16 Wl[2][128][32];   // [buf][m][c]
  __shared__ bf16 Xl[2][128][32];   // [buf][t][c] (transposed, masked)

  const int tid  = threadIdx.x;
  const int wv   = tid >> 5;     // wave 0..7  -> m-tile
  const int lane = tid & 31;
  const int r    = lane & 15;
  const int hf   = lane >> 4;

  const int mo = blockIdx.x * 128;
  const int no = blockIdx.y * 128;
  const int b  = blockIdx.z;

  const float* Xb = X + (size_t)b * E_DIM * T_DIM;
  const float* mb = mask + (size_t)b * T_DIM;

  v8f acc[8] = {};

  // staging thread mapping
  const int wm  = tid >> 1;          // W stage: row 0..127
  const int wc0 = (tid & 1) * 16;    // W stage: c base (4 quads)
  const int xc  = tid >> 3;          // X stage: c row 0..31
  const int xt0 = tid & 7;           // X stage: t quad base

  v4f wr[4], xr[4], mr[4];
  for (int j = 0; j < 4; ++j)
    mr[j] = *(const v4f*)(mb + no + (xt0 + j * 8) * 4);   // mask is kk-invariant

  auto load_regs = [&](int kk) {
    for (int j = 0; j < 4; ++j)
      wr[j] = *(const v4f*)(W + (size_t)(mo + wm) * E_DIM + kk + wc0 + j * 4);
    for (int j = 0; j < 4; ++j)
      xr[j] = *(const v4f*)(Xb + (size_t)(kk + xc) * T_DIM + no + (xt0 + j * 8) * 4);
  };
  auto store_lds = [&](int bi) {
    for (int j = 0; j < 4; ++j) {
      const int c = wc0 + j * 4;
      Wl[bi][wm][c + 0] = (bf16)wr[j].x;
      Wl[bi][wm][c + 1] = (bf16)wr[j].y;
      Wl[bi][wm][c + 2] = (bf16)wr[j].z;
      Wl[bi][wm][c + 3] = (bf16)wr[j].w;
    }
    for (int j = 0; j < 4; ++j) {
      const int t = (xt0 + j * 8) * 4;
      Xl[bi][t + 0][xc] = (bf16)(xr[j].x * mr[j].x);
      Xl[bi][t + 1][xc] = (bf16)(xr[j].y * mr[j].y);
      Xl[bi][t + 2][xc] = (bf16)(xr[j].z * mr[j].z);
      Xl[bi][t + 3][xc] = (bf16)(xr[j].w * mr[j].w);
    }
  };

  load_regs(0);
  store_lds(0);
  __syncthreads();

  for (int kk = 0; kk < E_DIM; kk += 32) {
    const int  cur  = (kk >> 5) & 1;
    const bool more = (kk + 32) < E_DIM;
    if (more) load_regs(kk + 32);          // overlap with WMMAs below
    if (kk + 64 < E_DIM) {                 // L2 prefetch hints (global_prefetch_b8)
      __builtin_prefetch(W + (size_t)(mo + wm) * E_DIM + kk + 64 + wc0, 0, 3);
      __builtin_prefetch(Xb + (size_t)(kk + 64 + xc) * T_DIM + no + xt0 * 4, 0, 3);
    }

    AFrag a;
    a.h[0] = *(const v8bf*)&Wl[cur][16 * wv + r][hf * 16];
    a.h[1] = *(const v8bf*)&Wl[cur][16 * wv + r][hf * 16 + 8];
    for (int nt = 0; nt < 8; ++nt) {
      v16bf bb = *(const v16bf*)&Xl[cur][nt * 16 + r][hf * 16];
      acc[nt] = wmma_bf16(a.v, bb, acc[nt]);
    }

    if (more) store_lds(cur ^ 1);
    __syncthreads();
  }

  const int o0 = mo + 16 * wv + 8 * hf;    // 8 consecutive output channels o0..o0+7
  v4f bA = *(const v4f*)(bias + o0);
  v4f bB = *(const v4f*)(bias + o0 + 4);
  float bv[8] = {bA.x, bA.y, bA.z, bA.w, bB.x, bB.y, bB.z, bB.w};

  if (MODE == 0 || MODE == 1) {
    const int h = o0 >> 6, d0 = o0 & 63;   // 8 channels never cross a head (d0 % 8 == 0)
    for (int nt = 0; nt < 8; ++nt) {
      const int t = no + nt * 16 + r;
      v8bf pk;
      for (int i = 0; i < 8; ++i) pk[i] = (bf16)(acc[nt][i] + bv[i]);
      *(v8bf*)(outB + (((size_t)b * H_DIM + h) * T_DIM + t) * DH_DIM + d0) = pk;
    }
  } else {
    for (int nt = 0; nt < 8; ++nt) {
      const int t = no + nt * 16 + r;
      for (int i = 0; i < 8; ++i) {
        const float val = acc[nt][i] + bv[i];
        if (MODE == 2)
          outB[((size_t)b * E_DIM + o0 + i) * T_DIM + t] = (bf16)val;
        else
          outF[((size_t)b * E_DIM + o0 + i) * T_DIM + t] = val;
      }
    }
  }
}

// ---------------------------------------------------------------------------
// Softmax stats: for each (b,h,q) compute m = max_k s and 1/sum_k exp(s-m).
// Stored packed per row as float4 (m, invL, qmask/32, 0) for one-b128 reads.
// s[q,k] = (qh . kh) * qmask[q] * kmask[k] / 32   (zeros participate, as ref)
// Per-lane online accumulation (lane owns a fixed column residue), single
// cross-lane flash-merge at the end -> no shuffles in the hot loop.
// ---------------------------------------------------------------------------
__global__ __launch_bounds__(128) void softmax_stats_kernel(
    const bf16* __restrict__ qp, const bf16* __restrict__ kp,
    const float* __restrict__ qmask, const float* __restrict__ kmask,
    v4f* __restrict__ stat4) {
  const int tid  = threadIdx.x;
  const int wv   = tid >> 5;
  const int lane = tid & 31;
  const int r    = lane & 15;
  const int hf   = lane >> 4;
  const int b = blockIdx.z, h = blockIdx.y;
  const int q0 = blockIdx.x * 64 + wv * 16;
  const size_t bhT = ((size_t)b * H_DIM + h) * T_DIM;

  const bf16* qrow = qp + (bhT + q0 + r) * DH_DIM;
  AFrag a1, a2;
  a1.h[0] = *(const v8bf*)(qrow + hf * 16);
  a1.h[1] = *(const v8bf*)(qrow + hf * 16 + 8);
  a2.h[0] = *(const v8bf*)(qrow + 32 + hf * 16);
  a2.h[1] = *(const v8bf*)(qrow + 32 + hf * 16 + 8);

  float qmf[8], mI[8], lI[8];
  for (int i = 0; i < 8; ++i) {
    qmf[i] = qmask[(size_t)b * T_DIM + q0 + i + 8 * hf] * 0.03125f;
    mI[i] = -3.0e38f;
    lI[i] = 0.0f;
  }

  for (int k0 = 0; k0 < T_DIM; k0 += 16) {
    const bf16* krow = kp + (bhT + k0 + r) * DH_DIM;
    v16bf b1 = *(const v16bf*)(krow + hf * 16);
    v16bf b2 = *(const v16bf*)(krow + 32 + hf * 16);
    v8f acc = {};
    acc = wmma_bf16(a1.v, b1, acc);
    acc = wmma_bf16(a2.v, b2, acc);
    const float km = kmask[(size_t)b * T_DIM + k0 + r];
    for (int i = 0; i < 8; ++i) {
      const float s  = acc[i] * qmf[i] * km;
      const float mn = fmaxf(mI[i], s);
      lI[i] = lI[i] * __expf(mI[i] - mn) + __expf(s - mn);
      mI[i] = mn;
    }
  }

  // merge the 16 per-lane (m,l) pairs within each half-wave
  for (int i = 0; i < 8; ++i) {
    float m = mI[i], l = lI[i];
    for (int off = 8; off >= 1; off >>= 1) {
      const float mo_ = __shfl_xor(m, off, 16);
      const float lo_ = __shfl_xor(l, off, 16);
      const float mn  = fmaxf(m, mo_);
      l = l * __expf(m - mn) + lo_ * __expf(mo_ - mn);
      m = mn;
    }
    if (r == 0) {
      stat4[bhT + q0 + i + 8 * hf] = (v4f){m, 1.0f / l, qmf[i], 0.0f};
    }
  }
}

// ---------------------------------------------------------------------------
// Attention output: O2[b, h*64+d, k] = sum_q vh[d,q] * p[q,k]
// p[q,k] = exp(s[q,k]*qm/32 - m[q]) * invL[q] * kmask[k], s via WMMA.
// Block: 256 threads (8 waves); block owns 128 k-columns (wave -> 16 cols),
// loops q in steps of 32 (one K=32 WMMA for the V@P accumulation).
// Row stats (m, invL, qm/32) staged once into LDS; transpose region is
// wave-private and DS ops are in-order per wave -> no barrier in the loop.
// ---------------------------------------------------------------------------
__global__ __launch_bounds__(256) void attn_out_kernel(
    const bf16* __restrict__ qp, const bf16* __restrict__ kp,
    const bf16* __restrict__ vp, const v4f* __restrict__ stat4,
    const float* __restrict__ kmask, float* __restrict__ O2) {
  __shared__ v4f  sSt[T_DIM];     // 32 KB: per-q (m, invL, qm/32, 0)
  __shared__ bf16 pl[8][16][32];  // per-wave region: [kcol 0..15][q' 0..31]

  const int tid  = threadIdx.x;
  const int wv   = tid >> 5;
  const int lane = tid & 31;
  const int r    = lane & 15;
  const int hf   = lane >> 4;
  const int b = blockIdx.z, h = blockIdx.y;
  const int k0 = blockIdx.x * 128 + wv * 16;
  const size_t bhT = ((size_t)b * H_DIM + h) * T_DIM;
  const int kcol = k0 + r;
  const float km = kmask[(size_t)b * T_DIM + kcol];

  // stage all row stats for this (b,h) into LDS once
  for (int j = 0; j < T_DIM / 256; ++j) sSt[tid + j * 256] = stat4[bhT + tid + j * 256];

  // B fragments of the score GEMM (fixed: this wave's 16 k-columns, d=0..63)
  const bf16* krow = kp + (bhT + kcol) * DH_DIM;
  v16bf bs1 = *(const v16bf*)(krow + hf * 16);
  v16bf bs2 = *(const v16bf*)(krow + 32 + hf * 16);

  v8f accO[4] = {};
  __syncthreads();

  for (int q0 = 0; q0 < T_DIM; q0 += 32) {
    for (int sub = 0; sub < 2; ++sub) {
      const bf16* qrow = qp + (bhT + q0 + sub * 16 + r) * DH_DIM;
      AFrag a1, a2;
      a1.h[0] = *(const v8bf*)(qrow + hf * 16);
      a1.h[1] = *(const v8bf*)(qrow + hf * 16 + 8);
      a2.h[0] = *(const v8bf*)(qrow + 32 + hf * 16);
      a2.h[1] = *(const v8bf*)(qrow + 32 + hf * 16 + 8);
      v8f s = {};
      s = wmma_bf16(a1.v, bs1, s);
      s = wmma_bf16(a2.v, bs2, s);
      for (int i = 0; i < 8; ++i) {
        const int q = q0 + sub * 16 + i + 8 * hf;
        const v4f st = sSt[q];                       // (m, invL, qm/32, 0)
        const float p = __expf(s[i] * st.z * km - st.x) * (st.y * km);
        pl[wv][r][sub * 16 + i + 8 * hf] = (bf16)p;  // transpose via LDS
      }
    }
    // wave-private region + in-order DS => safe without __syncthreads
    v16bf bp = *(const v16bf*)&pl[wv][r][hf * 16];
    for (int mt = 0; mt < 4; ++mt) {
      const bf16* vrow =
          vp + ((size_t)b * E_DIM + h * DH_DIM + mt * 16 + r) * T_DIM + q0;
      AFrag av;
      av.h[0] = *(const v8bf*)(vrow + hf * 16);
      av.h[1] = *(const v8bf*)(vrow + hf * 16 + 8);
      accO[mt] = wmma_bf16(av.v, bp, accO[mt]);
    }
  }

  for (int mt = 0; mt < 4; ++mt) {
    for (int i = 0; i < 8; ++i) {
      const int e = h * DH_DIM + mt * 16 + i + 8 * hf;
      O2[((size_t)b * E_DIM + e) * T_DIM + k0 + r] = accO[mt][i];
    }
  }
}

// ---------------------------------------------------------------------------
extern "C" void kernel_launch(void* const* d_in, const int* in_sizes, int n_in,
                              void* d_out, int out_size, void* d_ws,
                              size_t ws_size, hipStream_t stream) {
  const float* q     = (const float*)d_in[0];
  const float* k     = (const float*)d_in[1];
  const float* v     = (const float*)d_in[2];
  const float* qmask = (const float*)d_in[3];
  const float* kmask = (const float*)d_in[4];
  const float* vmask = (const float*)d_in[5];
  const float* Wq    = (const float*)d_in[6];
  const float* bq    = (const float*)d_in[7];
  const float* Wk    = (const float*)d_in[8];
  const float* bk    = (const float*)d_in[9];
  const float* Wv    = (const float*)d_in[10];
  const float* bv    = (const float*)d_in[11];
  const float* Wo    = (const float*)d_in[12];
  const float* bo    = (const float*)d_in[13];

  char* ws = (char*)d_ws;
  const size_t szP = (size_t)B_DIM * E_DIM * T_DIM * sizeof(bf16);  // 8 MB
  bf16* qp = (bf16*)(ws);                       // [B,H,T,DH]
  bf16* kp = (bf16*)(ws + szP);                 // [B,H,T,DH]
  bf16* vp = (bf16*)(ws + 2 * szP);             // [B,E,T]
  float* O2 = (float*)(ws + 3 * szP);           // [B,E,T] f32 (16 MB)
  v4f* stat4 = (v4f*)(ws + 3 * szP + (size_t)B_DIM * E_DIM * T_DIM * 4);

  dim3 gProj(E_DIM / 128, T_DIM / 128, B_DIM);
  proj_kernel<0><<<gProj, 256, 0, stream>>>(Wq, q, qmask, bq, qp, nullptr);
  proj_kernel<1><<<gProj, 256, 0, stream>>>(Wk, k, kmask, bk, kp, nullptr);
  proj_kernel<2><<<gProj, 256, 0, stream>>>(Wv, v, vmask, bv, vp, nullptr);

  softmax_stats_kernel<<<dim3(T_DIM / 64, H_DIM, B_DIM), 128, 0, stream>>>(
      qp, kp, qmask, kmask, stat4);

  attn_out_kernel<<<dim3(T_DIM / 128, H_DIM, B_DIM), 256, 0, stream>>>(
      qp, kp, vp, stat4, kmask, O2);

  proj_kernel<3><<<gProj, 256, 0, stream>>>(Wo, O2, kmask, bo, nullptr,
                                            (float*)d_out);
}
